// STMambaDKAN_30288109371863
// MI455X (gfx1250) — matrix operand order repaired and checked by reading, not compiled, not run on hardware
//
#include <hip/hip_runtime.h>

// ---------------- problem constants ----------------
constexpr int B_ = 8, T_ = 48, N_ = 2048, H_ = 64, P_ = 12;
constexpr int ROWS = B_ * T_ * N_;        // 786432
constexpr int BT = B_ * T_;               // 384

// ---------------- WMMA types / helpers ----------------
typedef __attribute__((ext_vector_type(16))) __bf16 v16bf;
typedef __attribute__((ext_vector_type(8)))  float  v8f;

union AFrag { v16bf v; unsigned int u[8]; };
union BFrag { v16bf v; uint4 q[2]; };
union CAcc  { v8f  v; float f[8]; };

__device__ __forceinline__ unsigned short f2bf(float x) {
  __bf16 h = (__bf16)x;
  return __builtin_bit_cast(unsigned short, h);
}
__device__ __forceinline__ unsigned int pack2(float lo, float hi) {
  return (unsigned int)f2bf(lo) | ((unsigned int)f2bf(hi) << 16);
}
__device__ __forceinline__ v8f wmma_bf16(v16bf a, v16bf b, v8f c) {
  // D = A(16x32 bf16) * B(32x16 bf16) + C(16x16 f32)
  return __builtin_amdgcn_wmma_f32_16x16x32_bf16(false, a, false, b, (short)0, c, false, false);
}
// fast sigmoid: v_exp + v_rcp (no IEEE div chains)
__device__ __forceinline__ float sigm(float x) {
  return __builtin_amdgcn_rcpf(1.0f + __expf(-x));
}
// tanh-form gelu rewritten via sigmoid: 0.5x(1+tanh(u)) == x*sigmoid(2u)
__device__ __forceinline__ float gelu_tanh(float x) {
  const float u2 = 1.5957691216057308f * (x + 0.044715f * x * x * x);
  return x * sigm(u2);
}

// A-fragment (16x32 bf16) from row-major bf16(ushort) LDS tile.
// layout: lane m = row; V0-7: K = 16*half + 2v + {0,1}
__device__ __forceinline__ v16bf load_afrag(const unsigned short* lds, int ld, int k0, int lane) {
  const int m = lane & 15, half = lane >> 4;
  AFrag f;
#pragma unroll
  for (int v = 0; v < 8; ++v) {
    const int koff = 16 * half + 2 * v;
    const unsigned short* p = lds + m * ld + k0 + koff;
    f.u[v] = (unsigned int)p[0] | ((unsigned int)p[1] << 16);
  }
  return f.v;
}

// A-fragment from a row-major *f32* LDS tile (converted at fragment-load time)
__device__ __forceinline__ v16bf load_afrag_f32(const float* lds, int ld, int k0, int lane) {
  const int m = lane & 15, half = lane >> 4;
  AFrag f;
#pragma unroll
  for (int v = 0; v < 8; ++v) {
    const int koff = 16 * half + 2 * v;
    const float2 p = *(const float2*)(lds + m * ld + k0 + koff);
    f.u[v] = pack2(p.x, p.y);
  }
  return f.v;
}

// B-fragment (32x16 bf16) from a column-major (transposed) LDS tile: ldsT[(col)*32 + k]
// layout used here (contiguous k): lane n = col; V0-7: K = 16*half + 2v + {0,1}
__device__ __forceinline__ v16bf load_bfragT(const unsigned short* ldsT, int ntbase, int lane) {
  const int n = lane & 15, half = lane >> 4;
  AFrag f;
#pragma unroll
  for (int v = 0; v < 8; ++v) {
    const int k = 16 * half + 2 * v;
    const unsigned short* p = ldsT + (ntbase + n) * 32 + k;
    f.u[v] = (unsigned int)p[0] | ((unsigned int)p[1] << 16);
  }
  return f.v;
}

// B-fragment from pre-packed LDS fragment buffer (8 uints per lane per frag)
__device__ __forceinline__ v16bf load_bfrag(const unsigned int* frag, int fragIdx, int lane) {
  const unsigned int* p = frag + fragIdx * 256 + lane * 8;
  BFrag f;
  f.q[0] = *(const uint4*)(p);
  f.q[1] = *(const uint4*)(p + 4);
  return f.v;
}

// Pre-pack weight W (Kin x Nout, f32 row-major) into B-fragment layout in LDS.
__device__ __forceinline__ void prepack_weight(const float* __restrict__ W, int Kin, int Nout,
                                               unsigned int* frag, int fragBase,
                                               int ksteps, int ntiles, int tid, int nthreads) {
  const int total = ksteps * ntiles * 32;
  for (int e = tid; e < total; e += nthreads) {
    const int lane = e & 31, f = e >> 5;
    const int ks = f / ntiles, nt = f - ks * ntiles;
    const int n = nt * 16 + (lane & 15), half = lane >> 4;
    unsigned int* dst = frag + (fragBase + f) * 256 + lane * 8;
#pragma unroll
    for (int v = 0; v < 8; ++v) {
      const int k = ks * 32 + 16 * half + 2 * v;
      float lo = 0.f, hi = 0.f;
      if (n < Nout) { lo = W[k * Nout + n]; hi = W[(k + 1) * Nout + n]; }
      dst[v] = pack2(lo, hi);
    }
  }
}

// ---------------- K1: emb = x@W_feat + b_feat + tod_emb[tidx] + node_emb ----------------
__global__ void __launch_bounds__(256) k1_embed(const float* __restrict__ x,
                                                const int* __restrict__ tidx,
                                                const float* __restrict__ Wf,
                                                const float* __restrict__ bf,
                                                const float* __restrict__ tod,
                                                const float* __restrict__ nemb,
                                                float* __restrict__ emb) {
  const int gid = blockIdx.x * 256 + threadIdx.x;  // over ROWS*H
  const int h = gid & 63;
  const int row = gid >> 6;
  const int bt = row >> 11;                        // row / N
  const float* xr = x + (long)row * 3;
  float acc = bf[h] + nemb[h];
  acc += xr[0] * Wf[0 * 64 + h] + xr[1] * Wf[1 * 64 + h] + xr[2] * Wf[2 * 64 + h];
  acc += tod[tidx[bt] * 64 + h];
  emb[(long)gid] = acc;
}

// ---------------- K2: fused expand + EMA-gate + W_con -> cat[...,0:64] ----------------
__global__ void __launch_bounds__(128) k2_temporal(const float* __restrict__ emb,
                                                   const float* __restrict__ Wexp,
                                                   const float* __restrict__ bexp,
                                                   const float* __restrict__ Wcon,
                                                   const float* __restrict__ bcon,
                                                   float* __restrict__ cat) {
  __shared__ __align__(16) unsigned int s_wf[24 * 256];   // 16 frags W_exp + 8 frags W_con
  __shared__ unsigned short s_x[4][16 * 64];
  __shared__ unsigned short s_p[4][16 * 64];
  const int tid = threadIdx.x, lane = tid & 31, w = tid >> 5;
  const int col = lane & 15, half = lane >> 4;
  prepack_weight(Wexp, 64, 128, s_wf, 0, 2, 8, tid, 128);
  prepack_weight(Wcon, 64, 64, s_wf, 16, 2, 4, tid, 128);
  __syncthreads();
  const int b = blockIdx.x >> 5;           // 32 node-tiles of 64
  const int node0 = (blockIdx.x & 31) * 64 + w * 16;

  float bp[4], bgv[4], bc[4];
#pragma unroll
  for (int nt = 0; nt < 4; ++nt) {
    bp[nt]  = bexp[nt * 16 + col];
    bgv[nt] = bexp[64 + nt * 16 + col];
    bc[nt]  = bcon[nt * 16 + col];
  }
  CAcc sst[4];                              // EMA state, lives in C-fragment layout
#pragma unroll
  for (int nt = 0; nt < 4; ++nt)
#pragma unroll
    for (int i = 0; i < 8; ++i) sst[nt].f[i] = 0.f;

  for (int t = 0; t < 48; ++t) {
    const float* xb = emb + ((long)(b * 48 + t) * 2048 + node0) * 64;
    for (int q = lane; q < 256; q += 32) {      // stage 16x64 f32 -> bf16 LDS
      const int r = q >> 4, c4 = (q & 15) << 2;
      const float4 xv = *(const float4*)(xb + r * 64 + c4);
      unsigned short* d = &s_x[w][r * 64 + c4];
      d[0] = f2bf(xv.x); d[1] = f2bf(xv.y); d[2] = f2bf(xv.z); d[3] = f2bf(xv.w);
    }
    CAcc pa[4], ga[4];
#pragma unroll
    for (int nt = 0; nt < 4; ++nt)
#pragma unroll
      for (int i = 0; i < 8; ++i) { pa[nt].f[i] = bp[nt]; ga[nt].f[i] = bgv[nt]; }
#pragma unroll
    for (int ks = 0; ks < 2; ++ks) {
      const v16bf a = load_afrag(s_x[w], 64, ks * 32, lane);
#pragma unroll
      for (int nt = 0; nt < 4; ++nt) pa[nt].v = wmma_bf16(a, load_bfrag(s_wf, ks * 8 + nt, lane), pa[nt].v);
#pragma unroll
      for (int nt = 0; nt < 4; ++nt) ga[nt].v = wmma_bf16(a, load_bfrag(s_wf, ks * 8 + 4 + nt, lane), ga[nt].v);
    }
#pragma unroll
    for (int nt = 0; nt < 4; ++nt)
#pragma unroll
      for (int i = 0; i < 8; ++i) {
        sst[nt].f[i] = 0.9f * sst[nt].f[i] + 0.1f * ga[nt].f[i];
        const float prod = pa[nt].f[i] * sigm(sst[nt].f[i]);
        s_p[w][(i + 8 * half) * 64 + nt * 16 + col] = f2bf(prod);
      }
    CAcc ca[4];
#pragma unroll
    for (int nt = 0; nt < 4; ++nt)
#pragma unroll
      for (int i = 0; i < 8; ++i) ca[nt].f[i] = bc[nt];
#pragma unroll
    for (int ks = 0; ks < 2; ++ks) {
      const v16bf a = load_afrag(s_p[w], 64, ks * 32, lane);
#pragma unroll
      for (int nt = 0; nt < 4; ++nt) ca[nt].v = wmma_bf16(a, load_bfrag(s_wf, 16 + ks * 4 + nt, lane), ca[nt].v);
    }
    const long rowb = (long)(b * 48 + t) * 2048 + node0;
#pragma unroll
    for (int nt = 0; nt < 4; ++nt)
#pragma unroll
      for (int i = 0; i < 8; ++i)
        cat[(rowb + i + 8 * half) * 128 + nt * 16 + col] = ca[nt].f[i];
  }
}

// ---------------- K3: agg = adj @ emb  (+ gelu(agg@W_spline+b)) -> cat[...,64:128] ----------------
__global__ void __launch_bounds__(128) k3_spatial(const float* __restrict__ adj,
                                                  const float* __restrict__ emb,
                                                  const float* __restrict__ Wsp,
                                                  const float* __restrict__ bsp,
                                                  float* __restrict__ cat) {
  __shared__ __align__(16) unsigned int s_wf[8 * 256];    // W_spline fragments
  __shared__ unsigned short s_embT[64 * 32];              // transposed emb tile (col-major)
  __shared__ __align__(16) float s_a[4][32 * 32];         // per-wave adj tile (f32, async-filled)
  __shared__ unsigned short s_c[4][32 * 64];              // per-wave C round-trip
  const int tid = threadIdx.x, lane = tid & 31, w = tid >> 5;
  const int col = lane & 15, half = lane >> 4;
  prepack_weight(Wsp, 64, 64, s_wf, 0, 2, 4, tid, 128);
  const int bt = blockIdx.x >> 4;           // (b*T+t)
  const int r0 = (blockIdx.x & 15) * 128 + w * 32;   // node row of this wave
  const float* embb = emb + (long)bt * (2048 * 64);
  const unsigned ldsbase = (unsigned)(uintptr_t)(&s_a[w][0]);

  CAcc acc[2][4];
#pragma unroll
  for (int m2 = 0; m2 < 2; ++m2)
#pragma unroll
    for (int nt = 0; nt < 4; ++nt)
#pragma unroll
      for (int i = 0; i < 8; ++i) acc[m2][nt].f[i] = 0.f;

  for (int kk = 0; kk < 64; ++kk) {
    const int k0 = kk * 32;
    __syncthreads();
    // ---- async copy adj tile (32 rows x 32 cols, f32) straight into LDS ----
#pragma unroll
    for (int it = 0; it < 8; ++it) {
      const int q = lane + it * 32;
      const int r = q >> 3, c4 = (q & 7) << 2;
      const unsigned voff  = (unsigned)(((r0 + r) * 2048 + k0 + c4) * 4);
      const unsigned laddr = ldsbase + (unsigned)((r * 32 + c4) * 4);
      asm volatile("global_load_async_to_lds_b128 %0, %1, %2"
                   :: "v"(laddr), "v"(voff), "s"(adj)
                   : "memory");
    }
    // ---- stage emb[k0..k0+31][0..63] transposed: ldsT[h*32+k] ----
    for (int idx = tid; idx < 512; idx += 128) {
      const int k = idx >> 4, h4 = (idx & 15) << 2;
      const float4 xv = *(const float4*)(embb + (long)(k0 + k) * 64 + h4);
      s_embT[(h4 + 0) * 32 + k] = f2bf(xv.x);
      s_embT[(h4 + 1) * 32 + k] = f2bf(xv.y);
      s_embT[(h4 + 2) * 32 + k] = f2bf(xv.z);
      s_embT[(h4 + 3) * 32 + k] = f2bf(xv.w);
    }
    asm volatile("s_wait_asynccnt 0x0" ::: "memory");
    __syncthreads();
    v16bf bfr[4];
#pragma unroll
    for (int nt = 0; nt < 4; ++nt) bfr[nt] = load_bfragT(s_embT, nt * 16, lane);
#pragma unroll
    for (int m2 = 0; m2 < 2; ++m2) {
      const v16bf a = load_afrag_f32(&s_a[w][m2 * 16 * 32], 32, 0, lane);
#pragma unroll
      for (int nt = 0; nt < 4; ++nt) acc[m2][nt].v = wmma_bf16(a, bfr[nt], acc[m2][nt].v);
    }
  }
  // ---- fused spline epilogue: spatial = gelu(C @ W_spline + b) ----
#pragma unroll
  for (int m2 = 0; m2 < 2; ++m2)
#pragma unroll
    for (int nt = 0; nt < 4; ++nt)
#pragma unroll
      for (int i = 0; i < 8; ++i)
        s_c[w][(m2 * 16 + i + 8 * half) * 64 + nt * 16 + col] = f2bf(acc[m2][nt].f[i]);
  float bv[4];
#pragma unroll
  for (int nt = 0; nt < 4; ++nt) bv[nt] = bsp[nt * 16 + col];
  CAcc sp[2][4];
#pragma unroll
  for (int m2 = 0; m2 < 2; ++m2)
#pragma unroll
    for (int nt = 0; nt < 4; ++nt)
#pragma unroll
      for (int i = 0; i < 8; ++i) sp[m2][nt].f[i] = bv[nt];
#pragma unroll
  for (int m2 = 0; m2 < 2; ++m2)
#pragma unroll
    for (int ks = 0; ks < 2; ++ks) {
      const v16bf a = load_afrag(&s_c[w][m2 * 16 * 64], 64, ks * 32, lane);
#pragma unroll
      for (int nt = 0; nt < 4; ++nt) sp[m2][nt].v = wmma_bf16(a, load_bfrag(s_wf, ks * 4 + nt, lane), sp[m2][nt].v);
    }
  const long rowbase = (long)bt * 2048 + r0;
#pragma unroll
  for (int m2 = 0; m2 < 2; ++m2)
#pragma unroll
    for (int nt = 0; nt < 4; ++nt)
#pragma unroll
      for (int i = 0; i < 8; ++i) {
        const long row = rowbase + m2 * 16 + i + 8 * half;
        cat[row * 128 + 64 + nt * 16 + col] = gelu_tanh(sp[m2][nt].f[i]);
      }
}

// ---------------- K4: gate/merge/fuse + head -> out ----------------
__global__ void __launch_bounds__(128) k4_final(const float* __restrict__ cat,
                                                const float* __restrict__ Wg, const float* __restrict__ bg,
                                                const float* __restrict__ Wm, const float* __restrict__ bm,
                                                const float* __restrict__ Wh, const float* __restrict__ bh,
                                                float* __restrict__ out) {
  __shared__ __align__(16) unsigned int s_wf[34 * 256];   // 16 gate + 16 merge + 2 head frags
  __shared__ unsigned short s_x[4][16 * 128];
  __shared__ unsigned short s_f[4][16 * 64];
  const int tid = threadIdx.x, lane = tid & 31, w = tid >> 5;
  const int col = lane & 15, half = lane >> 4;
  prepack_weight(Wg, 128, 64, s_wf, 0, 4, 4, tid, 128);
  prepack_weight(Wm, 128, 64, s_wf, 16, 4, 4, tid, 128);
  prepack_weight(Wh, 64, 12, s_wf, 32, 2, 1, tid, 128);
  __syncthreads();
  const long row0 = ((long)blockIdx.x * 4 + w) * 16;
  // stage 16x128 cat tile -> bf16 LDS
  for (int q = lane; q < 512; q += 32) {
    const int r = q >> 5, c4 = (q & 31) << 2;
    const float4 xv = *(const float4*)(cat + (row0 + r) * 128 + c4);
    unsigned short* d = &s_x[w][r * 128 + c4];
    d[0] = f2bf(xv.x); d[1] = f2bf(xv.y); d[2] = f2bf(xv.z); d[3] = f2bf(xv.w);
  }
  float bgv[4], bmv[4];
#pragma unroll
  for (int nt = 0; nt < 4; ++nt) { bgv[nt] = bg[nt * 16 + col]; bmv[nt] = bm[nt * 16 + col]; }
  CAcc ga[4], ma[4];
#pragma unroll
  for (int nt = 0; nt < 4; ++nt)
#pragma unroll
    for (int i = 0; i < 8; ++i) { ga[nt].f[i] = bgv[nt]; ma[nt].f[i] = bmv[nt]; }
#pragma unroll
  for (int ks = 0; ks < 4; ++ks) {
    const v16bf a = load_afrag(s_x[w], 128, ks * 32, lane);
#pragma unroll
    for (int nt = 0; nt < 4; ++nt) ga[nt].v = wmma_bf16(a, load_bfrag(s_wf, ks * 4 + nt, lane), ga[nt].v);
#pragma unroll
    for (int nt = 0; nt < 4; ++nt) ma[nt].v = wmma_bf16(a, load_bfrag(s_wf, 16 + ks * 4 + nt, lane), ma[nt].v);
  }
  // fused = g*merged + (1-g)*temporal_state  (temporal state reloaded in fp32)
#pragma unroll
  for (int nt = 0; nt < 4; ++nt)
#pragma unroll
    for (int i = 0; i < 8; ++i) {
      const long row = row0 + i + 8 * half;
      const float ts = cat[row * 128 + nt * 16 + col];
      const float g = sigm(ga[nt].f[i]);
      const float fused = g * ma[nt].f[i] + (1.0f - g) * ts;
      s_f[w][(i + 8 * half) * 64 + nt * 16 + col] = f2bf(fused);
    }
  CAcc hacc;
#pragma unroll
  for (int i = 0; i < 8; ++i) hacc.f[i] = (col < 12) ? bh[col < 12 ? col : 0] : 0.f;
#pragma unroll
  for (int ks = 0; ks < 2; ++ks) {
    const v16bf a = load_afrag(s_f[w], 64, ks * 32, lane);
    hacc.v = wmma_bf16(a, load_bfrag(s_wf, 32 + ks, lane), hacc.v);
  }
  if (col < 12) {
#pragma unroll
    for (int i = 0; i < 8; ++i)
      out[(row0 + i + 8 * half) * 12 + col] = hacc.f[i];
  }
}

// ---------------- host launcher ----------------
extern "C" void kernel_launch(void* const* d_in, const int* in_sizes, int n_in,
                              void* d_out, int out_size, void* d_ws, size_t ws_size,
                              hipStream_t stream) {
  const float* x      = (const float*)d_in[0];
  const float* adj    = (const float*)d_in[1];
  const int*   tidx   = (const int*)  d_in[2];
  const float* W_feat = (const float*)d_in[3];
  const float* b_feat = (const float*)d_in[4];
  const float* tod    = (const float*)d_in[5];
  const float* nemb   = (const float*)d_in[6];
  const float* W_exp  = (const float*)d_in[7];
  const float* b_exp  = (const float*)d_in[8];
  const float* W_con  = (const float*)d_in[9];
  const float* b_con  = (const float*)d_in[10];
  const float* W_spl  = (const float*)d_in[11];
  const float* b_spl  = (const float*)d_in[12];
  const float* W_gate = (const float*)d_in[13];
  const float* b_gate = (const float*)d_in[14];
  const float* W_mrg  = (const float*)d_in[15];
  const float* b_mrg  = (const float*)d_in[16];
  const float* W_head = (const float*)d_in[17];
  const float* b_head = (const float*)d_in[18];
  float* out = (float*)d_out;

  float* emb = (float*)d_ws;                       // ROWS*H f32   (201 MB)
  float* cat = emb + (size_t)ROWS * H_;            // ROWS*2H f32  (403 MB)

  k1_embed   <<<(ROWS * H_) / 256, 256, 0, stream>>>(x, tidx, W_feat, b_feat, tod, nemb, emb);
  k2_temporal<<<B_ * (N_ / 64),   128, 0, stream>>>(emb, W_exp, b_exp, W_con, b_con, cat);
  k3_spatial <<<BT * (N_ / 128),  128, 0, stream>>>(adj, emb, W_spl, b_spl, cat);
  k4_final   <<<(ROWS / 16) / 4,  128, 0, stream>>>(cat, W_gate, b_gate, W_mrg, b_mrg,
                                                    W_head, b_head, out);
  (void)in_sizes; (void)n_in; (void)out_size; (void)ws_size;
}
